// CausalRecurrenceLayer_77163382440504
// MI455X (gfx1250) — matrix operand order repaired and checked by reading, not compiled, not run on hardware
//
#include <hip/hip_runtime.h>
#include <hip/hip_bf16.h>

// ---------------- problem constants (from reference setup) ----------------
#define BB   4
#define LL   4096
#define DD   1024
#define NN   (BB * LL)          // 16384 tokens
#define CH   32                 // scan chunks per sequence
#define CL   (LL / CH)          // 128 steps per chunk

// GEMM tiling
#define KCHUNK   128            // K elems staged per TDM transfer
#define NROWS    64             // tokens per block (4 n-tiles per wave)
#define NCHUNKS  (DD / KCHUNK)  // 8
#define LDS_HALF (NROWS * KCHUNK)          // bf16 elems per buffer (8192)
#define LDS_BYTES (2 * LDS_HALF * 2)       // double buffered: 32 KB

typedef __attribute__((ext_vector_type(16))) __bf16    v16bf;
typedef __attribute__((ext_vector_type(8)))  float     v8f;
typedef __attribute__((ext_vector_type(4)))  unsigned  v4u;
typedef __attribute__((ext_vector_type(8)))  int       v8i;
typedef __attribute__((ext_vector_type(4)))  int       v4i;

union Frag { v16bf bf; v4u u[2]; };

static __device__ __forceinline__ __bf16 f2bf(float f) {
    unsigned u = __builtin_bit_cast(unsigned, f);
    unsigned r = (u + 0x7FFFu + ((u >> 16) & 1u)) >> 16;     // RNE
    return __builtin_bit_cast(__bf16, (unsigned short)r);
}
static __device__ __forceinline__ float sigmoidf(float x) {
    return 1.0f / (1.0f + __expf(-x));
}

// ---- Tensor Data Mover: load a 2D bf16 tile (KCHUNK x NROWS, row stride DD
// elems) from global into packed LDS rows of KCHUNK elems. ISA 8.3/8.4 D#.
// clang-23 toolchain: 6-arg builtin (g0 u32x4, g1 i32x8, g2 i32x4, g3 i32x4,
// extra i32x8, cpol i32).
static __device__ __forceinline__ void tdm_load_tile(unsigned lds_off_bytes,
                                                     const void* gptr) {
    unsigned long long ga = (unsigned long long)gptr;
    v4u g0;
    g0.x = 1u;                                        // count=1, user descriptor
    g0.y = lds_off_bytes;                             // lds_addr
    g0.z = (unsigned)(ga & 0xFFFFFFFFull);            // global_addr[31:0]
    g0.w = (unsigned)((ga >> 32) & 0x01FFFFFFull)     // global_addr[56:32]
         | 0x80000000u;                               // type = 2 ("image")
    v8i g1;
    // tensor_dim0 = DD elems: bits79:48 -> w1[31:16] low16, w2[15:0] high16
    g1[0] = 0x00010000;                               // data_size=1 (2B), mask=0
    g1[1] = (int)((DD & 0xFFFF) << 16);
    g1[2] = (int)((0xFFFFu) << 16) | (DD >> 16);      // tensor_dim1=0xFFFF (ample)
    g1[3] = (int)(KCHUNK << 16);                      // tile_dim0=128 (w3[31:16])
    g1[4] = NROWS;                                    // tile_dim1=64, tile_dim2=0
    g1[5] = DD;                                       // tensor_dim0_stride low32
    g1[6] = 0;
    g1[7] = 0;
    v4i z4 = {0, 0, 0, 0};
    v8i z8 = {0, 0, 0, 0, 0, 0, 0, 0};
    __builtin_amdgcn_tensor_load_to_lds(g0, g1, z4, z4, z8, 0);
}

// ---------------- kernel 1: weight conversion + per-channel decay const ----
__global__ void k_prep(const float* __restrict__ Wr, const float* __restrict__ Wi,
                       const float* __restrict__ Wo, const float* __restrict__ log_a,
                       __bf16* __restrict__ wr, __bf16* __restrict__ wi,
                       __bf16* __restrict__ wo, float* __restrict__ l2a8) {
    int t = blockIdx.x * blockDim.x + threadIdx.x;
    if (t < DD * DD) {
        wr[t] = f2bf(Wr[t]);
        wi[t] = f2bf(Wi[t]);
        wo[t] = f2bf(Wo[t]);
    }
    if (t < DD) {
        float ab = sigmoidf(log_a[t]);               // a_base
        l2a8[t] = 8.0f * __log2f(ab);                // a_t = exp2(r * l2a8)
    }
}

// ---------------- kernel 2: causal depthwise conv (k=4, left pad 3) -------
__global__ void k_conv(const float* __restrict__ x, const float* __restrict__ cw,
                       const float* __restrict__ cb,
                       float* __restrict__ xcf, __bf16* __restrict__ xcb) {
    int t = blockIdx.x * blockDim.x + threadIdx.x;   // over N*D
    if (t >= NN * DD) return;
    int j = t & (DD - 1);
    int l = (t >> 10) & (LL - 1);
    int b = t >> 22;
    const float* w = cw + j * 4;
    float acc = cb[j];
    #pragma unroll
    for (int k = 0; k < 4; ++k) {
        int ll = l - 3 + k;
        if (ll >= 0) acc += w[k] * x[((size_t)(b * LL + ll)) * DD + j];
    }
    xcf[t] = acc;
    xcb[t] = f2bf(acc);
}

// ---------------- kernel 3: fused dual GEMM (W_r, W_i) + gate epilogue ----
// TDM double-buffers 64-token x 128-K A chunks into LDS; 8 waves each own one
// 16-channel j-tile and 4 n-tiles (B fragments reused 4x). A fragments are
// batch-loaded per K-step (one DS clause, one wait), then 8 WMMAs issue
// back-to-back.
__global__ void __launch_bounds__(256)
k_gates(const __bf16* __restrict__ A, const __bf16* __restrict__ wr,
        const __bf16* __restrict__ wi, const float* __restrict__ br,
        const float* __restrict__ bi, const float* __restrict__ l2a8,
        const float* __restrict__ xcf, float* __restrict__ a_buf,
        float* __restrict__ b_buf) {
    extern __shared__ __bf16 smem[];                 // 2 x 64 x 128 bf16
    const int lane = threadIdx.x & 31;
    const int wv   = threadIdx.x >> 5;
    const int nb   = blockIdx.x >> 3;                // 256 n-blocks
    const int jb   = blockIdx.x & 7;                 // 8 j-blocks
    const int n0   = nb * NROWS;
    const int j0   = jb * 128 + wv * 16;
    const int half = lane >> 4;
    const int lrow = lane & 15;

    const __bf16* pBr = wr + (size_t)(j0 + lrow) * DD + half * 8;
    const __bf16* pBi = wi + (size_t)(j0 + lrow) * DD + half * 8;
    const __bf16* gA  = A  + (size_t)n0 * DD;        // 64-row strip base

    v8f accR[4] = {{}, {}, {}, {}};
    v8f accI[4] = {{}, {}, {}, {}};

    if (wv == 0) tdm_load_tile(0u, gA);
    for (int c = 0; c < NCHUNKS; ++c) {
        if (wv == 0) __builtin_amdgcn_s_wait_tensorcnt(0);
        __syncthreads();                              // chunk c visible; buf (c+1)&1 free
        if (wv == 0 && c + 1 < NCHUNKS)
            tdm_load_tile((unsigned)(((c + 1) & 1) * LDS_HALF * 2),
                          (const char*)gA + (c + 1) * (KCHUNK * 2));
        const __bf16* sb = smem + (c & 1) * LDS_HALF;
        #pragma unroll
        for (int ks = 0; ks < KCHUNK / 32; ++ks) {
            const int kg = c * KCHUNK + ks * 32;
            Frag fr, fi;
            fr.u[0] = *(const v4u*)(pBr + kg);
            fr.u[1] = *(const v4u*)(pBr + kg + 16);
            fi.u[0] = *(const v4u*)(pBi + kg);
            fi.u[1] = *(const v4u*)(pBi + kg + 16);
            Frag a[4];
            #pragma unroll
            for (int nt = 0; nt < 4; ++nt) {          // batch all DS loads
                const __bf16* lp = sb + (nt * 16 + lrow) * KCHUNK + ks * 32 + half * 8;
                a[nt].u[0] = *(const v4u*)lp;
                a[nt].u[1] = *(const v4u*)(lp + 16);
            }
            #pragma unroll
            for (int nt = 0; nt < 4; ++nt) {          // then 8 back-to-back WMMAs
                accR[nt] = __builtin_amdgcn_wmma_f32_16x16x32_bf16(
                    false, a[nt].bf, false, fr.bf, (short)0, accR[nt], false, false);
                accI[nt] = __builtin_amdgcn_wmma_f32_16x16x32_bf16(
                    false, a[nt].bf, false, fi.bf, (short)0, accI[nt], false, false);
            }
        }
        __syncthreads();                              // done reading buf c&1
    }

    const int j = j0 + lrow;
    const float bR = br[j], bI = bi[j], la = l2a8[j];
    #pragma unroll
    for (int nt = 0; nt < 4; ++nt) {
        #pragma unroll
        for (int e = 0; e < 8; ++e) {
            int n = n0 + nt * 16 + e + half * 8;
            size_t idx = (size_t)n * DD + j;
            float rt = sigmoidf(accR[nt][e] + bR);
            float it = sigmoidf(accI[nt][e] + bI);
            float at = exp2f(rt * la);
            float sc = sqrtf(fmaxf(0.0f, 1.0f - at * at));
            float xc = xcf[idx];                      // read before aliased write
            a_buf[idx] = at;
            b_buf[idx] = sc * it * xc;                // b_buf aliases xcf (same elem)
        }
    }
}

// ---------------- kernels 4-6: chunked parallel scan ----------------------
__global__ void k_scan1(const float* __restrict__ a_buf, const float* __restrict__ b_buf,
                        float* __restrict__ aggA, float* __restrict__ aggB) {
    int t = blockIdx.x * blockDim.x + threadIdx.x;   // over B*CH*D
    if (t >= BB * CH * DD) return;
    int c = t & (DD - 1);
    int ch = (t >> 10) & (CH - 1);
    int b = t >> 15;
    size_t base = ((size_t)b * LL + ch * CL) * DD + c;
    float Ap = 1.0f, Bp = 0.0f;
    for (int s = 0; s < CL; ++s) {
        float a = a_buf[base + (size_t)s * DD];
        float v = b_buf[base + (size_t)s * DD];
        Bp = a * Bp + v;
        Ap *= a;
    }
    aggA[t] = Ap;
    aggB[t] = Bp;
}

__global__ void k_scan2(const float* __restrict__ aggA, const float* __restrict__ aggB,
                        float* __restrict__ hin) {
    int t = blockIdx.x * blockDim.x + threadIdx.x;   // over B*D
    if (t >= BB * DD) return;
    int c = t & (DD - 1);
    int b = t >> 10;
    float h = 0.0f;
    for (int ch = 0; ch < CH; ++ch) {
        int idx = ((b * CH) + ch) * DD + c;
        hin[idx] = h;
        h = aggA[idx] * h + aggB[idx];
    }
}

__global__ void k_scan3(const float* __restrict__ a_buf, const float* __restrict__ b_buf,
                        const float* __restrict__ hin, __bf16* __restrict__ hs) {
    int t = blockIdx.x * blockDim.x + threadIdx.x;   // over B*CH*D
    if (t >= BB * CH * DD) return;
    int c = t & (DD - 1);
    int ch = (t >> 10) & (CH - 1);
    int b = t >> 15;
    size_t base = ((size_t)b * LL + ch * CL) * DD + c;
    float h = hin[t];
    for (int s = 0; s < CL; ++s) {
        float a = a_buf[base + (size_t)s * DD];
        float v = b_buf[base + (size_t)s * DD];
        h = a * h + v;
        hs[base + (size_t)s * DD] = f2bf(h);
    }
}

// ---------------- kernel 7: output GEMM (hs @ W_out^T), same TDM pipeline --
__global__ void __launch_bounds__(256)
k_out(const __bf16* __restrict__ A, const __bf16* __restrict__ wo,
      float* __restrict__ y) {
    extern __shared__ __bf16 smem[];
    const int lane = threadIdx.x & 31;
    const int wv   = threadIdx.x >> 5;
    const int nb   = blockIdx.x >> 3;
    const int jb   = blockIdx.x & 7;
    const int n0   = nb * NROWS;
    const int j0   = jb * 128 + wv * 16;
    const int half = lane >> 4;
    const int lrow = lane & 15;

    const __bf16* pB = wo + (size_t)(j0 + lrow) * DD + half * 8;
    const __bf16* gA = A  + (size_t)n0 * DD;

    v8f acc[4] = {{}, {}, {}, {}};

    if (wv == 0) tdm_load_tile(0u, gA);
    for (int c = 0; c < NCHUNKS; ++c) {
        if (wv == 0) __builtin_amdgcn_s_wait_tensorcnt(0);
        __syncthreads();
        if (wv == 0 && c + 1 < NCHUNKS)
            tdm_load_tile((unsigned)(((c + 1) & 1) * LDS_HALF * 2),
                          (const char*)gA + (c + 1) * (KCHUNK * 2));
        const __bf16* sb = smem + (c & 1) * LDS_HALF;
        #pragma unroll
        for (int ks = 0; ks < KCHUNK / 32; ++ks) {
            const int kg = c * KCHUNK + ks * 32;
            Frag fb;
            fb.u[0] = *(const v4u*)(pB + kg);
            fb.u[1] = *(const v4u*)(pB + kg + 16);
            Frag a[4];
            #pragma unroll
            for (int nt = 0; nt < 4; ++nt) {
                const __bf16* lp = sb + (nt * 16 + lrow) * KCHUNK + ks * 32 + half * 8;
                a[nt].u[0] = *(const v4u*)lp;
                a[nt].u[1] = *(const v4u*)(lp + 16);
            }
            #pragma unroll
            for (int nt = 0; nt < 4; ++nt) {
                acc[nt] = __builtin_amdgcn_wmma_f32_16x16x32_bf16(
                    false, a[nt].bf, false, fb.bf, (short)0, acc[nt], false, false);
            }
        }
        __syncthreads();
    }

    const int j = j0 + lrow;
    #pragma unroll
    for (int nt = 0; nt < 4; ++nt) {
        #pragma unroll
        for (int e = 0; e < 8; ++e) {
            int n = n0 + nt * 16 + e + half * 8;
            y[(size_t)n * DD + j] = acc[nt][e];
        }
    }
}

// ---------------- kernel 8: RMSNorm (one wave per token row) --------------
__global__ void k_rms(const float* __restrict__ y, const float* __restrict__ gamma,
                      float* __restrict__ out) {
    const int lane = threadIdx.x & 31;
    const int row  = blockIdx.x * 8 + (threadIdx.x >> 5);
    const float* p = y + (size_t)row * DD;
    float s = 0.0f;
    #pragma unroll
    for (int t = 0; t < DD / 32; ++t) {
        float v = p[lane + 32 * t];
        s += v * v;
    }
    #pragma unroll
    for (int off = 16; off > 0; off >>= 1)
        s += __shfl_xor(s, off, 32);
    float rms = rsqrtf(s * (1.0f / DD) + 1e-6f);
    float* o = out + (size_t)row * DD;
    #pragma unroll
    for (int t = 0; t < DD / 32; ++t) {
        int j = lane + 32 * t;
        o[j] = p[j] * rms * gamma[j];
    }
}

// ---------------- host launcher -------------------------------------------
extern "C" void kernel_launch(void* const* d_in, const int* /*in_sizes*/, int /*n_in*/,
                              void* d_out, int /*out_size*/, void* d_ws, size_t /*ws_size*/,
                              hipStream_t stream) {
    const float* x      = (const float*)d_in[0];
    const float* conv_w = (const float*)d_in[1];
    const float* conv_b = (const float*)d_in[2];
    const float* W_r    = (const float*)d_in[3];
    const float* b_r    = (const float*)d_in[4];
    const float* W_i    = (const float*)d_in[5];
    const float* b_i    = (const float*)d_in[6];
    const float* log_a  = (const float*)d_in[7];
    const float* W_out  = (const float*)d_in[8];
    const float* gamma  = (const float*)d_in[9];
    float* out = (float*)d_out;

    char* ws = (char*)d_ws;
    size_t off = 0;
    auto alloc = [&](size_t bytes) { char* p = ws + off; off += (bytes + 255) & ~(size_t)255; return p; };

    float*  xcf  = (float*) alloc((size_t)NN * DD * 4);   // x_conv fp32, later b_t
    float*  a_y  = (float*) alloc((size_t)NN * DD * 4);   // a_t, later y (pre-norm)
    __bf16* xcb  = (__bf16*)alloc((size_t)NN * DD * 2);   // x_conv bf16, later hs bf16
    __bf16* wr   = (__bf16*)alloc((size_t)DD * DD * 2);
    __bf16* wi   = (__bf16*)alloc((size_t)DD * DD * 2);
    __bf16* wo   = (__bf16*)alloc((size_t)DD * DD * 2);
    float*  l2a8 = (float*) alloc((size_t)DD * 4);
    float*  aggA = (float*) alloc((size_t)BB * CH * DD * 4);
    float*  aggB = (float*) alloc((size_t)BB * CH * DD * 4);
    float*  hin  = (float*) alloc((size_t)BB * CH * DD * 4);

    k_prep <<<(DD * DD + 255) / 256, 256, 0, stream>>>(W_r, W_i, W_out, log_a, wr, wi, wo, l2a8);
    k_conv <<<((size_t)NN * DD + 255) / 256, 256, 0, stream>>>(x, conv_w, conv_b, xcf, xcb);

    // 256 token-blocks (64 rows) x 8 channel-blocks (8 waves x 16 ch)
    k_gates<<<(NN / NROWS) * (DD / 128), 256, LDS_BYTES, stream>>>(
        xcb, wr, wi, b_r, b_i, l2a8, xcf, a_y, /*b_buf=*/xcf);

    k_scan1<<<(BB * CH * DD) / 256, 256, 0, stream>>>(a_y, xcf, aggA, aggB);
    k_scan2<<<(BB * DD) / 256, 256, 0, stream>>>(aggA, aggB, hin);
    k_scan3<<<(BB * CH * DD) / 256, 256, 0, stream>>>(a_y, xcf, hin, /*hs=*/xcb);

    k_out  <<<(NN / NROWS) * (DD / 128), 256, LDS_BYTES, stream>>>(xcb, wo, /*y=*/a_y);
    k_rms  <<<NN / 8, 256, 0, stream>>>(a_y, gamma, out);
}
